// GCN_56307021250665
// MI455X (gfx1250) — compile-verified, matrix operand order checked by reading, not run on hardware
//
#include <hip/hip_runtime.h>
#include <hip/hip_bf16.h>
#include <stdint.h>

#define NNODES 100000
#define NEDGES 640000
#define D_IN   128
#define D_H    128
#define D_OUT  64
#define BN_EPS 1e-5f

typedef __attribute__((ext_vector_type(16))) __bf16 v16bf;
typedef __attribute__((ext_vector_type(8)))  float  v8f;

static __device__ __forceinline__ unsigned short f32_to_bf16_rne(float f) {
    union { float f; uint32_t u; } v; v.f = f;
    uint32_t u = v.u;
    uint32_t r = u + 0x7FFFu + ((u >> 16) & 1u);
    if ((u & 0x7F800000u) == 0x7F800000u) r = u;  // Inf/NaN: truncate
    return (unsigned short)(r >> 16);
}

// ---------------- degree / norm ----------------
__global__ void k_fill_f32(float* __restrict__ p, float val, int n) {
    int t = blockIdx.x * blockDim.x + threadIdx.x;
    if (t < n) p[t] = val;
}

__global__ void k_deg_accum(const long long* __restrict__ dst, float* __restrict__ deg, int e) {
    int t = blockIdx.x * blockDim.x + threadIdx.x;
    if (t < e) atomicAdd(&deg[(int)dst[t]], 1.0f);
}

__global__ void k_rsqrt_inplace(float* __restrict__ p, int n) {
    int t = blockIdx.x * blockDim.x + threadIdx.x;
    if (t < n) p[t] = rsqrtf(p[t]);
}

// ---------------- weight transpose + bf16 cast: W[K,F] -> Wt[F,K] ----------------
__global__ void k_wt_cast(const float* __restrict__ W, unsigned short* __restrict__ Wt,
                          int K, int F) {
    int t = blockIdx.x * blockDim.x + threadIdx.x;
    if (t < K * F) {
        int k = t / F, f = t - k * F;
        Wt[f * K + k] = f32_to_bf16_rne(W[t]);
    }
}

// ---------------- BatchNorm (eval) + bf16 cast; feature dim = 128 ----------------
__global__ void k_bn_cast(const float* __restrict__ in,
                          const float* __restrict__ g, const float* __restrict__ b,
                          const float* __restrict__ m, const float* __restrict__ v,
                          unsigned short* __restrict__ out, int total) {
    int t = blockIdx.x * blockDim.x + threadIdx.x;
    if (t < total) {
        int f = t & 127;
        float sc = g[f] * rsqrtf(v[f] + BN_EPS);
        out[t] = f32_to_bf16_rne((in[t] - m[f]) * sc + b[f]);
    }
}

// ---------------- GEMM: C[N,F] = A(bf16)[N,128] x Wt(bf16)[F,128]^T, f32 accum ----------------
template <int F>
__global__ void k_gemm_wmma(const unsigned short* __restrict__ A,
                            const unsigned short* __restrict__ Wt,
                            float* __restrict__ C, int nrows) {
    __shared__ __align__(32) unsigned short sW[F * 128];
    {   // cooperative stage of the whole weight tile into LDS (b128 copies)
        const uint4* ws = (const uint4*)Wt;
        uint4*       wd = (uint4*)sW;
        const int nv = F * 128 * 2 / 16;
        for (int i = threadIdx.x; i < nv; i += blockDim.x) wd[i] = ws[i];
    }
    __syncthreads();

    const int lane = threadIdx.x & 31;
    const int wave = blockIdx.x * (blockDim.x >> 5) + (threadIdx.x >> 5);
    const int m0   = wave * 16;                 // N % 16 == 0, no partial tiles
    if (m0 >= nrows) return;                    // uniform per-wave, EXEC stays all-1s

    const int lrow = lane & 15;
    const int lhi  = lane >> 4;

    // A fragments: lane holds row m0+lrow, K-half lhi*16, 4 K-steps of 32
    const unsigned short* arow = A + (size_t)(m0 + lrow) * 128 + lhi * 16;
    v16bf af[4];
#pragma unroll
    for (int kt = 0; kt < 4; ++kt)
        af[kt] = *(const v16bf*)(arow + kt * 32);

#pragma unroll
    for (int nt = 0; nt < (F >> 4); ++nt) {
        const unsigned short* brow = sW + (size_t)(nt * 16 + lrow) * 128 + lhi * 16;
        v8f acc = {};
#pragma unroll
        for (int kt = 0; kt < 4; ++kt) {
            v16bf bfrag = *(const v16bf*)(brow + kt * 32);   // ds_load_b128 x2
            acc = __builtin_amdgcn_wmma_f32_16x16x32_bf16(
                      false, af[kt], false, bfrag, (short)0, acc, false, false);
        }
        // C layout: VGPR v -> row m0 + v + 8*lhi, col nt*16 + lrow
        float* cb = C + (size_t)(m0 + lhi * 8) * F + nt * 16 + lrow;
#pragma unroll
        for (int v = 0; v < 8; ++v) cb[(size_t)v * F] = acc[v];
    }
}

// ---------------- self-loop init: agg[i,f] = hW[i,f] * dinv[i]^2 ----------------
template <int F>
__global__ void k_self(const float* __restrict__ hw, const float* __restrict__ dinv,
                       float* __restrict__ agg, int total) {
    int t = blockIdx.x * blockDim.x + threadIdx.x;
    if (t < total) {
        int i = t / F;
        float d = dinv[i];
        agg[t] = hw[t] * d * d;
    }
}

// ---------------- edge scatter: one wave per edge, float4 per lane ----------------
template <int F>
__global__ void k_edge(const long long* __restrict__ src, const long long* __restrict__ dst,
                       const float* __restrict__ dinv, const float* __restrict__ hw,
                       float* __restrict__ agg) {
    int t = blockIdx.x * blockDim.x + threadIdx.x;
    int e = t >> 5, lane = t & 31;
    if (e >= NEDGES) return;
    int s = (int)src[e], d = (int)dst[e];
    float coef = dinv[s] * dinv[d];
    const float4* h4 = (const float4*)(hw + (size_t)s * F);
    float*        a  = agg + (size_t)d * F;
    constexpr int Q = F / 4;
#pragma unroll
    for (int q = lane; q < Q; q += 32) {
        float4 hv = h4[q];                       // coalesced 512B gather per edge
        atomicAdd(&a[4 * q + 0], hv.x * coef);   // global_atomic_add_f32 (L2-resident)
        atomicAdd(&a[4 * q + 1], hv.y * coef);
        atomicAdd(&a[4 * q + 2], hv.z * coef);
        atomicAdd(&a[4 * q + 3], hv.w * coef);
    }
}

// ---------------- finalize: += bias, optional ReLU (in place) ----------------
template <int F, bool RELU>
__global__ void k_fin(float* __restrict__ agg, const float* __restrict__ bias, int total) {
    int t = blockIdx.x * blockDim.x + threadIdx.x;
    if (t < total) {
        float v = agg[t] + bias[t % F];
        agg[t] = RELU ? fmaxf(v, 0.0f) : v;
    }
}

extern "C" void kernel_launch(void* const* d_in, const int* in_sizes, int n_in,
                              void* d_out, int out_size, void* d_ws, size_t ws_size,
                              hipStream_t stream) {
    const float*     x    = (const float*)d_in[0];
    const long long* ei   = (const long long*)d_in[1];   // int64 [2,E]
    const long long* src  = ei;
    const long long* dstv = ei + NEDGES;
    const float* W1 = (const float*)d_in[4];  const float* b1 = (const float*)d_in[5];
    const float* Wm = (const float*)d_in[6];  const float* bm = (const float*)d_in[7];
    const float* W2 = (const float*)d_in[8];  const float* b2 = (const float*)d_in[9];
    const float* bn1g = (const float*)d_in[10]; const float* bn1b = (const float*)d_in[11];
    const float* bn1m = (const float*)d_in[12]; const float* bn1v = (const float*)d_in[13];
    const float* bnmg = (const float*)d_in[14]; const float* bnmb = (const float*)d_in[15];
    const float* bnmm = (const float*)d_in[16]; const float* bnmv = (const float*)d_in[17];
    const float* bn2g = (const float*)d_in[18]; const float* bn2b = (const float*)d_in[19];
    const float* bn2m = (const float*)d_in[20]; const float* bn2v = (const float*)d_in[21];

    // outputs (tuple flat): out [N,64], res1 [N,128], res2 [N,128]
    float* out0 = (float*)d_out;
    float* res1 = out0 + (size_t)NNODES * D_OUT;
    float* res2 = res1 + (size_t)NNODES * D_H;

    // workspace carve-out
    size_t off = 0;
    auto alloc = [&](size_t bytes) -> void* {
        void* p = (char*)d_ws + off;
        off += (bytes + 255) & ~(size_t)255;
        return p;
    };
    float*          dinv = (float*)alloc(sizeof(float) * NNODES);
    unsigned short* hbf  = (unsigned short*)alloc(sizeof(unsigned short) * (size_t)NNODES * 128);
    float*          hw   = (float*)alloc(sizeof(float) * (size_t)NNODES * 128);
    unsigned short* wt1  = (unsigned short*)alloc(sizeof(unsigned short) * 128 * 128);
    unsigned short* wtm  = (unsigned short*)alloc(sizeof(unsigned short) * 128 * 128);
    unsigned short* wt2  = (unsigned short*)alloc(sizeof(unsigned short) * 64 * 128);

    const int TB = 256;
    auto nb = [](long long n) { return (int)((n + 255) / 256); };

    // degrees -> dinv
    k_fill_f32<<<nb(NNODES), TB, 0, stream>>>(dinv, 1.0f, NNODES);
    k_deg_accum<<<nb(NEDGES), TB, 0, stream>>>(dstv, dinv, NEDGES);
    k_rsqrt_inplace<<<nb(NNODES), TB, 0, stream>>>(dinv, NNODES);

    // weights -> transposed bf16
    k_wt_cast<<<nb(128 * 128), TB, 0, stream>>>(W1, wt1, 128, 128);
    k_wt_cast<<<nb(128 * 128), TB, 0, stream>>>(Wm, wtm, 128, 128);
    k_wt_cast<<<nb(128 * 64),  TB, 0, stream>>>(W2, wt2, 128, 64);

    const long long totH = (long long)NNODES * 128;
    const long long totO = (long long)NNODES * 64;
    const int gemmBlocks = ((NNODES / 16) + 7) / 8;          // 8 waves/block, 16 rows/wave
    const int edgeBlocks = nb((long long)NEDGES * 32);

    // ---- layer 1: res1 = relu(GCN(BN1(x), W1) + b1)
    k_bn_cast<<<nb(totH), TB, 0, stream>>>(x, bn1g, bn1b, bn1m, bn1v, hbf, (int)totH);
    k_gemm_wmma<128><<<gemmBlocks, TB, 0, stream>>>(hbf, wt1, hw, NNODES);
    k_self<128><<<nb(totH), TB, 0, stream>>>(hw, dinv, res1, (int)totH);
    k_edge<128><<<edgeBlocks, TB, 0, stream>>>(src, dstv, dinv, hw, res1);
    k_fin<128, true><<<nb(totH), TB, 0, stream>>>(res1, b1, (int)totH);

    // ---- layer 2: res2 = relu(GCN(BNm(res1), Wm) + bm)
    k_bn_cast<<<nb(totH), TB, 0, stream>>>(res1, bnmg, bnmb, bnmm, bnmv, hbf, (int)totH);
    k_gemm_wmma<128><<<gemmBlocks, TB, 0, stream>>>(hbf, wtm, hw, NNODES);
    k_self<128><<<nb(totH), TB, 0, stream>>>(hw, dinv, res2, (int)totH);
    k_edge<128><<<edgeBlocks, TB, 0, stream>>>(src, dstv, dinv, hw, res2);
    k_fin<128, true><<<nb(totH), TB, 0, stream>>>(res2, bm, (int)totH);

    // ---- layer 3: out = GCN(BN2(res2), W2) + b2   (no relu)
    k_bn_cast<<<nb(totH), TB, 0, stream>>>(res2, bn2g, bn2b, bn2m, bn2v, hbf, (int)totH);
    k_gemm_wmma<64><<<gemmBlocks, TB, 0, stream>>>(hbf, wt2, hw, NNODES);
    k_self<64><<<nb(totO), TB, 0, stream>>>(hw, dinv, out0, (int)totO);
    k_edge<64><<<edgeBlocks, TB, 0, stream>>>(src, dstv, dinv, hw, out0);
    k_fin<64, false><<<nb(totO), TB, 0, stream>>>(out0, b2, (int)totO);
}
